// SignalPredictorAllocator_1872605741359
// MI455X (gfx1250) — matrix operand order repaired and checked by reading, not compiled, not run on hardware
//
#include <hip/hip_runtime.h>
#include <math.h>

typedef __attribute__((ext_vector_type(16))) __bf16 v16bf;
typedef __attribute__((ext_vector_type(8)))  float  v8f;

#define BATCH      2048
#define NASSET     512
#define DIM        128
#define HID        256
#define M_TOTAL    (BATCH*NASSET)          // 1048576 rows
#define M_BLOCK    64
#define LDS_STRIDE 132                      // 128 + 4 pad: bank-conflict-free column reads

// ---------------------------------------------------------------------------
// Kernel 0: split W1 (fp32 [128,256]) into hi/lo bf16, pre-packed in WMMA
// B-fragment order.  Fragment F = c*16 + T (c = K-chunk of 32, T = 16-col tile).
// Within a fragment: lane L holds column n = 16T + (L&15); its 16 elements are
// K = 32c + 16*(L>>4) + i  (i = 0..15), the same k order as the A fragment.
// Storage: element (F, L, i) at flat index F*512 + L*16 + i  -> each lane's
// fragment is a contiguous, 32B-aligned v16bf.
// ---------------------------------------------------------------------------
__global__ void repack_w1(const float* __restrict__ W1,
                          __bf16* __restrict__ hi, __bf16* __restrict__ lo) {
  int t = blockIdx.x * blockDim.x + threadIdx.x;   // 32768 threads total
  int F = t >> 9;
  int L = (t >> 4) & 31;
  int i = t & 15;
  int c = F >> 4;
  int T = F & 15;
  int n = T * 16 + (L & 15);
  int k = c * 32 + (L >> 4) * 16 + i;
  float w = W1[k * HID + n];
  __bf16 h = (__bf16)w;
  hi[t] = h;
  lo[t] = (__bf16)(w - (float)h);
}

// ---------------------------------------------------------------------------
// Kernel 1: fused MLP.  Block = 256 threads = 8 waves (4 along M x 2 along H).
// The 64x128 fp32 A block (contiguous 32 KB in global) is staged to LDS with
// gfx1250 async memory->LDS DMA (GLOBAL_LOAD_ASYNC_TO_LDS_B128, ASYNCcnt),
// tagged non-temporal since signal_features is a 512 MB single-use stream.
// Each wave computes 16 rows x 128 cols as 8 WMMA tiles with bf16x3 split
// precision, then fuses bias, ReLU, the [256,1] second layer (LDS float
// atomics), b2 and sigmoid.
// ---------------------------------------------------------------------------
__global__ void __launch_bounds__(256) mlp_kernel(
    const float* __restrict__ A,
    const __bf16* __restrict__ Bh, const __bf16* __restrict__ Bl,
    const float* __restrict__ b1, const float* __restrict__ W2,
    const float* __restrict__ b2, float* __restrict__ ls) {
  __shared__ float sA[M_BLOCK * LDS_STRIDE];
  __shared__ float sScore[M_BLOCK];

  int tid  = threadIdx.x;
  int lane = tid & 31;
  int wave = tid >> 5;
  int wm   = wave & 3;     // M-tile of this wave (16 rows)
  int wh   = wave >> 2;    // H-half of this wave (128 cols)
  int hf   = lane >> 4;    // lane half (ISA A/B layout split)
  int l15  = lane & 15;

  if (tid < M_BLOCK) sScore[tid] = 0.f;

  // Async-stage the 64x128 fp32 A block into padded LDS.
  // LDS byte offset = low 32 bits of the generic shared-pointer (AMDGPU
  // addrspacecast AS3->flat: aperture in [63:32], LDS offset in [31:0]).
  const float4* g4 = (const float4*)(A + (size_t)blockIdx.x * (M_BLOCK * DIM));
#pragma unroll
  for (int p = 0; p < 8; ++p) {
    int q = tid + 256 * p;                  // 2048 float4 total
    int row = q >> 5, col = (q & 31) * 4;   // 32 float4 per row
    unsigned ldsoff = (unsigned)(size_t)&sA[row * LDS_STRIDE + col];
    unsigned long long gaddr = (unsigned long long)(size_t)(g4 + q);
    asm volatile("global_load_async_to_lds_b128 %0, %1, off th:TH_LOAD_NT"
                 :: "v"(ldsoff), "v"(gaddr) : "memory");
  }
  asm volatile("s_wait_asynccnt 0x0" ::: "memory");
  __syncthreads();

  v8f zero = {};
  v8f acc[8];
#pragma unroll
  for (int t = 0; t < 8; ++t) acc[t] = zero;

  int mrow = wm * 16 + l15;
  for (int c = 0; c < 4; ++c) {
    int kc = c * 32;
    // A fragment: element s -> K = kc + 16*hf + s (s = 0..15)
    const float* ap = &sA[mrow * LDS_STRIDE + kc + 16 * hf];
    v16bf ah = {}, al = {};
#pragma unroll
    for (int i = 0; i < 8; ++i) {
      float f0 = ap[i];
      float f1 = ap[8 + i];
      __bf16 h0 = (__bf16)f0, h1 = (__bf16)f1;
      ah[i] = h0;      ah[8 + i] = h1;
      al[i] = (__bf16)(f0 - (float)h0);
      al[8 + i] = (__bf16)(f1 - (float)h1);
    }
#pragma unroll
    for (int t = 0; t < 8; ++t) {
      int F = c * 16 + (wh * 8 + t);
      v16bf bh = *(const v16bf*)(Bh + (size_t)F * 512 + lane * 16);
      v16bf bl = *(const v16bf*)(Bl + (size_t)F * 512 + lane * 16);
      // bf16x3: hi*hi + lo*hi + hi*lo, fp32 accumulate (~2^-17 rel error)
      acc[t] = __builtin_amdgcn_wmma_f32_16x16x32_bf16(false, ah, false, bh,
                                                       (short)0, acc[t], false, false);
      acc[t] = __builtin_amdgcn_wmma_f32_16x16x32_bf16(false, al, false, bh,
                                                       (short)0, acc[t], false, false);
      acc[t] = __builtin_amdgcn_wmma_f32_16x16x32_bf16(false, ah, false, bl,
                                                       (short)0, acc[t], false, false);
    }
  }

  // Fused epilogue: bias + ReLU + dot with W2, reduce over H into sScore.
  float part[8];
#pragma unroll
  for (int j = 0; j < 8; ++j) part[j] = 0.f;
#pragma unroll
  for (int t = 0; t < 8; ++t) {
    int n = wh * 128 + t * 16 + l15;        // this lane's column in tile t
    float bias = b1[n];
    float w2v  = W2[n];
#pragma unroll
    for (int j = 0; j < 8; ++j) {           // acc elem j -> row m = 8*hf + j
      float v = acc[t][j] + bias;
      v = v > 0.f ? v : 0.f;
      part[j] += v * w2v;
    }
  }
#pragma unroll
  for (int j = 0; j < 8; ++j)
    atomicAdd(&sScore[wm * 16 + hf * 8 + j], part[j]);
  __syncthreads();

  if (tid < M_BLOCK) {
    float sc  = sScore[tid] + b2[0];
    float sig = 1.f / (1.f + __expf(-sc));
    ls[(size_t)blockIdx.x * M_BLOCK + tid] = sig - 0.5f;
  }
}

// ---------------------------------------------------------------------------
// Kernel 2: one wave32 per batch row.  16 assets per lane (idx = j*32+lane).
// Exact top-k semantics: iterative argmax with lowest-index tie-break, picked
// entries excluded via bitmask (so they can't win -inf ties again).
// ---------------------------------------------------------------------------
__global__ void __launch_bounds__(256) select_kernel(
    const float* __restrict__ spread, const float* __restrict__ vol,
    const float* __restrict__ ls, float* __restrict__ out) {
  const float NEG_INF = -__builtin_inff();
  int lane = threadIdx.x & 31;
  int row  = blockIdx.x * 8 + (threadIdx.x >> 5);

  const float* sp = spread + (size_t)row * NASSET;
  const float* vo = vol    + (size_t)row * NASSET;
  const float* lr = ls     + (size_t)row * NASSET;

  float ratio[16], lsv[16];
  int validAny = 0;
#pragma unroll
  for (int j = 0; j < 16; ++j) {
    int idx = j * 32 + lane;
    float s = sp[idx];
    float v = vo[idx];
    lsv[j] = lr[idx];
    int valid = (__builtin_isfinite(s) && s > 0.f) ? 1 : 0;
    validAny |= valid;
    ratio[j] = valid ? v / (s + 1e-8f) : NEG_INF;
  }
#pragma unroll
  for (int m = 16; m > 0; m >>= 1) validAny |= __shfl_xor(validAny, m, 32);

  // top-50 of ratio -> universe mask
  unsigned picked = 0, univ = 0;
  for (int it = 0; it < 50; ++it) {
    float bv = NEG_INF; int bi = 0x7fffffff;
#pragma unroll
    for (int j = 0; j < 16; ++j) {
      if (!((picked >> j) & 1)) {
        float v = ratio[j]; int idx = j * 32 + lane;
        if (v > bv || (v == bv && idx < bi)) { bv = v; bi = idx; }
      }
    }
#pragma unroll
    for (int m = 16; m > 0; m >>= 1) {
      float ov = __shfl_xor(bv, m, 32);
      int   oi = __shfl_xor(bi, m, 32);
      if (ov > bv || (ov == bv && oi < bi)) { bv = ov; bi = oi; }
    }
    if (lane == (bi & 31)) { picked |= 1u << (bi >> 5); univ |= 1u << (bi >> 5); }
  }
  if (!validAny) univ = 0xFFFFu;   // no valid spread -> whole universe allowed

  float mabs[16];
#pragma unroll
  for (int j = 0; j < 16; ++j)
    mabs[j] = ((univ >> j) & 1) ? fabsf(lsv[j]) : NEG_INF;

  // top-10 of |score| within universe -> trade mask; conf = global max
  unsigned picked2 = 0, sel = 0;
  float conf = NEG_INF;
  for (int it = 0; it < 10; ++it) {
    float bv = NEG_INF; int bi = 0x7fffffff;
#pragma unroll
    for (int j = 0; j < 16; ++j) {
      if (!((picked2 >> j) & 1)) {
        float v = mabs[j]; int idx = j * 32 + lane;
        if (v > bv || (v == bv && idx < bi)) { bv = v; bi = idx; }
      }
    }
#pragma unroll
    for (int m = 16; m > 0; m >>= 1) {
      float ov = __shfl_xor(bv, m, 32);
      int   oi = __shfl_xor(bi, m, 32);
      if (ov > bv || (ov == bv && oi < bi)) { bv = ov; bi = oi; }
    }
    if (it == 0) conf = bv;
    if (lane == (bi & 31)) { picked2 |= 1u << (bi >> 5); sel |= 1u << (bi >> 5); }
  }
  if (!__builtin_isfinite(conf)) conf = 0.f;

  float asum = 0.f;
#pragma unroll
  for (int j = 0; j < 16; ++j)
    if ((sel >> j) & 1) asum += fabsf(lsv[j]);
#pragma unroll
  for (int m = 16; m > 0; m >>= 1) asum += __shfl_xor(asum, m, 32);

  float inv  = 1.f / (asum + 1e-8f);
  float take = (conf >= 0.05f) ? 1.f : 0.f;
#pragma unroll
  for (int j = 0; j < 16; ++j) {
    int idx = j * 32 + lane;
    float a = ((sel >> j) & 1) ? lsv[j] * inv * take : 0.f;
    out[(size_t)row * NASSET + idx] = a;
  }
  if (lane == 0) out[(size_t)M_TOTAL + row] = conf;
}

// ---------------------------------------------------------------------------
extern "C" void kernel_launch(void* const* d_in, const int* in_sizes, int n_in,
                              void* d_out, int out_size, void* d_ws, size_t ws_size,
                              hipStream_t stream) {
  const float* A      = (const float*)d_in[0];  // [B,N,D]
  const float* spread = (const float*)d_in[1];  // [B,N]
  const float* vol    = (const float*)d_in[2];  // [B,N]
  const float* W1     = (const float*)d_in[3];  // [D,H]
  const float* b1     = (const float*)d_in[4];  // [H]
  const float* W2     = (const float*)d_in[5];  // [H,1]
  const float* b2     = (const float*)d_in[6];  // [1]
  float* out = (float*)d_out;                   // action [B,N] ++ confidence [B]

  // workspace: ls_score (4 MB) | W1 hi bf16 packed (64 KB) | W1 lo (64 KB)
  char*   ws  = (char*)d_ws;
  float*  lsw = (float*)ws;
  __bf16* w1h = (__bf16*)(ws + (size_t)M_TOTAL * sizeof(float));
  __bf16* w1l = (__bf16*)(ws + (size_t)M_TOTAL * sizeof(float) + 65536);

  repack_w1<<<DIM * HID / 256, 256, 0, stream>>>(W1, w1h, w1l);
  mlp_kernel<<<M_TOTAL / M_BLOCK, 256, 0, stream>>>(A, w1h, w1l, b1, W2, b2, lsw);
  select_kernel<<<BATCH / 8, 256, 0, stream>>>(spread, vol, lsw, out);
}
